// EncoderLayer_86852828659792
// MI455X (gfx1250) — matrix operand order repaired and checked by reading, not compiled, not run on hardware
//
#include <hip/hip_runtime.h>
#include <hip/hip_bf16.h>

// ---------------------------------------------------------------------------
// EncoderLayer (6-relation attention + FFN) for MI455X / gfx1250, wave32.
// f16 WMMA (v_wmma_f32_16x16x32_f16) for all GEMMs, f32 reductions.
// Mask traffic (201 MB) dominates the roofline -> softmax path fully
// vectorized (int4 mask loads, float4 LDS, packed f16 stores).
// ---------------------------------------------------------------------------

typedef __attribute__((ext_vector_type(16))) _Float16 v16h;
typedef __attribute__((ext_vector_type(8)))  _Float16 v8h;
typedef __attribute__((ext_vector_type(4)))  _Float16 v4h;
typedef __attribute__((ext_vector_type(8)))  float    v8f;

#define BB 32
#define SS 512
#define DD 256
#define HH 64
#define FF 256
#define RR 6

// ---- WMMA wrapper ---------------------------------------------------------
__device__ __forceinline__ v8f wmma_f16(v16h a, v16h b, v8f c) {
  return __builtin_amdgcn_wmma_f32_16x16x32_f16(
      /*neg_a=*/false, a, /*neg_b=*/false, b,
      /*c_mod=*/(short)0, c, /*reuse_a=*/false, /*reuse_b=*/false);
}

// ---- Fragment loaders (ISA 7.12.2 layouts, wave32) ------------------------
// A: 16x32 (MxK) f16 fragment. base -> [m0][k0], row-major, leading dim ld.
// Lane l: m = l&15, khalf = l>>4; elems 0..7 -> K = khalf*8+e, 8..15 -> +16.
__device__ __forceinline__ v16h load_a_frag(const _Float16* base, int ld, int lane) {
  int m = lane & 15, kh = lane >> 4;
  const _Float16* p = base + (size_t)m * ld + kh * 8;
  v8h lo = *(const v8h*)p;
  v8h hi = *(const v8h*)(p + 16);
  return __builtin_shufflevector(lo, hi, 0,1,2,3,4,5,6,7,8,9,10,11,12,13,14,15);
}

// B: 32x16 (KxN) f16 fragment. base -> [k0][n0]; lane l: n = l&15, khalf = l>>4;
// elems 0..7 -> K = khalf*8+e, 8..15 -> +16.
__device__ __forceinline__ v16h load_b_frag(const _Float16* base, int ld, int lane) {
  int n = lane & 15, kh = lane >> 4;
  v16h b;
#pragma unroll
  for (int s = 0; s < 16; ++s) {
    int k = (s < 8) ? kh * 8 + s : 16 + kh * 8 + (s - 8);
    b[s] = base[(size_t)k * ld + n];
  }
  return b;
}

// ---- f32 -> f16 conversion (4-wide) ---------------------------------------
__global__ void cvt_f32_f16_v4(const float4* __restrict__ in,
                               v4h* __restrict__ out, int n4) {
  int i = blockIdx.x * blockDim.x + threadIdx.x;
  if (i < n4) {
    float4 v = in[i];
    v4h h;
    h[0] = (_Float16)v.x; h[1] = (_Float16)v.y;
    h[2] = (_Float16)v.z; h[3] = (_Float16)v.w;
    out[i] = h;
  }
}

// ---- Stage 1: QKV projection ---------------------------------------------
// grid.x = B*R*(S/16), grid.y = 0..2 (Q, K^T, V). One wave per block.
// Q,V stored [B*R, S, H]; K stored transposed [B*R, H, S].
__global__ __launch_bounds__(32) void qkv_kernel(
    const _Float16* __restrict__ xh,
    const _Float16* __restrict__ wq, const _Float16* __restrict__ wk,
    const _Float16* __restrict__ wv,
    _Float16* __restrict__ Q, _Float16* __restrict__ Kt, _Float16* __restrict__ V) {
  int lane = threadIdx.x;
  int blk  = blockIdx.x;
  int which = blockIdx.y;            // 0=Q, 1=Kt, 2=V
  int sblk = blk & 31;               // S/16 = 32 row blocks
  int br   = blk >> 5;               // b*6 + r
  int r    = br % RR;
  int b    = br / RR;

  const _Float16* xb = xh + ((size_t)b * SS + sblk * 16) * DD;
  const _Float16* W  = (which == 0 ? wq : which == 1 ? wk : wv) + (size_t)r * DD * HH;

  v16h a[8];
#pragma unroll
  for (int k = 0; k < 8; ++k) a[k] = load_a_frag(xb + k * 32, DD, lane);

  int mbase = (lane >> 4) * 8;
  int ncol  = lane & 15;
#pragma unroll
  for (int n0 = 0; n0 < HH; n0 += 16) {
    v8f c = {};
#pragma unroll
    for (int k = 0; k < 8; ++k)
      c = wmma_f16(a[k], load_b_frag(W + (size_t)(k * 32) * HH + n0, HH, lane), c);

    int hcol = n0 + ncol;
    if (which == 1) {
      // Transposed store: the 8 D elements of this lane are consecutive s-rows
      // -> one contiguous 16-byte store.
      v8h pk;
#pragma unroll
      for (int i = 0; i < 8; ++i) pk[i] = (_Float16)c[i];
      *(v8h*)&Kt[((size_t)br * HH + hcol) * SS + sblk * 16 + mbase] = pk;
    } else {
      _Float16* dst = (which == 0) ? Q : V;
#pragma unroll
      for (int i = 0; i < 8; ++i) {
        int srow = sblk * 16 + i + mbase;
        dst[((size_t)br * SS + srow) * HH + hcol] = (_Float16)c[i];
      }
    }
  }
}

// ---- Stage 2: fused masked attention -------------------------------------
// grid.x = B*R*(S/16). One wave per block; 16 query rows vs full S=512 keys.
__global__ __launch_bounds__(32) void attn_kernel(
    const _Float16* __restrict__ Q, const _Float16* __restrict__ Kt,
    const _Float16* __restrict__ V, const int* __restrict__ mask,
    _Float16* __restrict__ ctx) {
  __shared__ float    sc[16 * SS];      // 32 KB scores (f32)
  __shared__ _Float16 ph[16 * SS];      // 16 KB unnormalized weights (f16)
  __shared__ float    recip[16];

  int lane = threadIdx.x;
  int blk  = blockIdx.x;
  int sblk = blk & 31;
  int br   = blk >> 5;

  const _Float16* Qb  = Q  + ((size_t)br * SS + sblk * 16) * HH;
  const _Float16* Ktb = Kt + (size_t)br * HH * SS;
  const _Float16* Vb  = V  + (size_t)br * SS * HH;
  const int*      mb  = mask + ((size_t)br * SS + sblk * 16) * SS;

  int mbase = (lane >> 4) * 8;
  int ncol  = lane & 15;

  // scores = Q (16x64) x K^T (64x512), K-dim split 0/32
  v16h a0 = load_a_frag(Qb + 0,  HH, lane);
  v16h a1 = load_a_frag(Qb + 32, HH, lane);
#pragma unroll 4
  for (int t0 = 0; t0 < SS; t0 += 16) {
    v8f c = {};
    c = wmma_f16(a0, load_b_frag(Ktb + t0,           SS, lane), c);
    c = wmma_f16(a1, load_b_frag(Ktb + 32 * SS + t0, SS, lane), c);
#pragma unroll
    for (int i = 0; i < 8; ++i) sc[(i + mbase) * SS + t0 + ncol] = c[i];
  }
  __syncthreads();

  // masked softmax; 2 lanes per row, 256 cols each; fully vectorized.
  int row = lane >> 1, half = lane & 1;
  {
    const int4* mr4 = (const int4*)(mb + (size_t)row * SS + half * 256);
    float4*     s4  = (float4*)(sc + row * SS + half * 256);
    float mx = -3.0e38f;
#pragma unroll 4
    for (int t = 0; t < 64; ++t) {
      int4   mk = mr4[t];                 // 16B global load (mask: hot path)
      float4 sv = s4[t];                  // 16B LDS load
      float v0 = mk.x ? -1.0e9f : sv.x * 0.125f;   // 1/sqrt(64)
      float v1 = mk.y ? -1.0e9f : sv.y * 0.125f;
      float v2 = mk.z ? -1.0e9f : sv.z * 0.125f;
      float v3 = mk.w ? -1.0e9f : sv.w * 0.125f;
      s4[t] = make_float4(v0, v1, v2, v3);
      mx = fmaxf(mx, fmaxf(fmaxf(v0, v1), fmaxf(v2, v3)));
    }
    mx = fmaxf(mx, __shfl_xor(mx, 1, 32));

    float sum = 0.f;
    v4h* p4 = (v4h*)(ph + row * SS + half * 256);
#pragma unroll 4
    for (int t = 0; t < 64; ++t) {
      float4 v = s4[t];
      float p0 = __expf(v.x - mx), p1 = __expf(v.y - mx);
      float p2 = __expf(v.z - mx), p3 = __expf(v.w - mx);
      sum += (p0 + p1) + (p2 + p3);
      v4h pk;                              // post-softmax zeroing of masked
      pk[0] = (_Float16)((v.x < -0.5e9f) ? 0.f : p0);
      pk[1] = (_Float16)((v.y < -0.5e9f) ? 0.f : p1);
      pk[2] = (_Float16)((v.z < -0.5e9f) ? 0.f : p2);
      pk[3] = (_Float16)((v.w < -0.5e9f) ? 0.f : p3);
      p4[t] = pk;                          // 8B LDS store
    }
    sum += __shfl_xor(sum, 1, 32);
    if (half == 0) recip[row] = 1.f / sum;
  }
  __syncthreads();

  // ctx = attn (16x512) x V (512x64); fold 1/denominator into D rows.
#pragma unroll
  for (int h0 = 0; h0 < HH; h0 += 16) {
    v8f c = {};
#pragma unroll 4
    for (int k0 = 0; k0 < SS; k0 += 32) {
      v16h a = load_a_frag(ph + k0, SS, lane);
      v16h b = load_b_frag(Vb + (size_t)k0 * HH + h0, HH, lane);
      c = wmma_f16(a, b, c);
    }
#pragma unroll
    for (int i = 0; i < 8; ++i) {
      int m = i + mbase;
      ctx[((size_t)br * SS + sblk * 16 + m) * HH + h0 + ncol] =
          (_Float16)(c[i] * recip[m]);
    }
  }
}

// ---- Stage 3: proj + residual + per-branch LN + branch sum ---------------
// grid.x = B*(S/16). One wave per block.
__global__ __launch_bounds__(32) void proj_kernel(
    const _Float16* __restrict__ ctx, const _Float16* __restrict__ wfch,
    const float* __restrict__ x, float* __restrict__ s32,
    _Float16* __restrict__ sh) {
  __shared__ float acc[16 * DD];   // 16 KB: proj + x for one branch
  __shared__ float ssum[16 * DD];  // 16 KB: running sum of LN'd branches

  int lane = threadIdx.x;
  int blk  = blockIdx.x;
  int sblk = blk & 31;
  int b    = blk >> 5;

  for (int i = lane; i < 16 * DD; i += 32) ssum[i] = 0.f;

  const float* xb = x + ((size_t)b * SS + sblk * 16) * DD;
  int mbase = (lane >> 4) * 8;
  int ncol  = lane & 15;
  int row = lane >> 1, half = lane & 1;

  for (int r = 0; r < RR; ++r) {
    const _Float16* cb = ctx + (((size_t)(b * RR + r)) * SS + sblk * 16) * HH;
    const _Float16* W  = wfch + (size_t)r * HH * DD;
    v16h a0 = load_a_frag(cb + 0,  HH, lane);
    v16h a1 = load_a_frag(cb + 32, HH, lane);
    __syncthreads();  // previous iteration finished reading acc
#pragma unroll 4
    for (int n0 = 0; n0 < DD; n0 += 16) {
      v8f c = {};
      c = wmma_f16(a0, load_b_frag(W + n0,           DD, lane), c);
      c = wmma_f16(a1, load_b_frag(W + 32 * DD + n0, DD, lane), c);
#pragma unroll
      for (int i = 0; i < 8; ++i) {
        int m = i + mbase;
        int d = n0 + ncol;
        acc[m * DD + d] = c[i] + xb[(size_t)m * DD + d];
      }
    }
    __syncthreads();
    // LayerNorm per row (gamma=1, beta=0), accumulate into ssum (float4).
    float4* ar4 = (float4*)(acc + row * DD + half * 128);
    float4* sr4 = (float4*)(ssum + row * DD + half * 128);
    float mean = 0.f, m2 = 0.f;
#pragma unroll 4
    for (int t = 0; t < 32; ++t) {
      float4 v = ar4[t];
      mean += (v.x + v.y) + (v.z + v.w);
    }
    mean += __shfl_xor(mean, 1, 32);
    mean *= (1.f / 256.f);
#pragma unroll 4
    for (int t = 0; t < 32; ++t) {
      float4 v = ar4[t];
      float d0 = v.x - mean, d1 = v.y - mean, d2 = v.z - mean, d3 = v.w - mean;
      m2 += (d0 * d0 + d1 * d1) + (d2 * d2 + d3 * d3);
    }
    m2 += __shfl_xor(m2, 1, 32);
    float inv = rsqrtf(m2 * (1.f / 256.f) + 1e-5f);
#pragma unroll 4
    for (int t = 0; t < 32; ++t) {
      float4 v = ar4[t];
      float4 s = sr4[t];
      s.x += (v.x - mean) * inv; s.y += (v.y - mean) * inv;
      s.z += (v.z - mean) * inv; s.w += (v.w - mean) * inv;
      sr4[t] = s;
    }
  }
  __syncthreads();

  float4* so4 = (float4*)(s32 + ((size_t)b * SS + sblk * 16) * DD);
  v4h*    sh4 = (v4h*)(sh + ((size_t)b * SS + sblk * 16) * DD);
  float4* ss4 = (float4*)ssum;
  for (int i = lane; i < 16 * DD / 4; i += 32) {
    float4 v = ss4[i];
    so4[i] = v;
    v4h h;
    h[0] = (_Float16)v.x; h[1] = (_Float16)v.y;
    h[2] = (_Float16)v.z; h[3] = (_Float16)v.w;
    sh4[i] = h;
  }
}

// ---- Stage 4: FFN + residual + final LN ----------------------------------
// grid.x = B*(S/16). One wave per block.
__global__ __launch_bounds__(32) void ffn_kernel(
    const _Float16* __restrict__ sh, const _Float16* __restrict__ w1h,
    const _Float16* __restrict__ w2h, const float* __restrict__ s32,
    float* __restrict__ out) {
  __shared__ _Float16 h1[16 * FF];  // 8 KB: relu(s@W1) in f16
  __shared__ float    z[16 * DD];   // 16 KB: ffn + residual

  int lane = threadIdx.x;
  int blk  = blockIdx.x;
  int sblk = blk & 31;
  int b    = blk >> 5;

  const _Float16* sb = sh + ((size_t)b * SS + sblk * 16) * DD;
  int mbase = (lane >> 4) * 8;
  int ncol  = lane & 15;

  v16h a[8];
#pragma unroll
  for (int k = 0; k < 8; ++k) a[k] = load_a_frag(sb + k * 32, DD, lane);

#pragma unroll 2
  for (int n0 = 0; n0 < FF; n0 += 16) {
    v8f c = {};
#pragma unroll
    for (int k = 0; k < 8; ++k)
      c = wmma_f16(a[k], load_b_frag(w1h + (size_t)(k * 32) * FF + n0, FF, lane), c);
#pragma unroll
    for (int i = 0; i < 8; ++i)
      h1[(i + mbase) * FF + n0 + ncol] = (_Float16)fmaxf(c[i], 0.f);
  }
  __syncthreads();

  const float* s32b = s32 + ((size_t)b * SS + sblk * 16) * DD;
#pragma unroll 2
  for (int n0 = 0; n0 < DD; n0 += 16) {
    v8f c = {};
#pragma unroll
    for (int k = 0; k < 8; ++k)
      c = wmma_f16(load_a_frag(h1 + k * 32, FF, lane),
                   load_b_frag(w2h + (size_t)(k * 32) * DD + n0, DD, lane), c);
#pragma unroll
    for (int i = 0; i < 8; ++i) {
      int m = i + mbase;
      int d = n0 + ncol;
      z[m * DD + d] = c[i] + s32b[(size_t)m * DD + d];
    }
  }
  __syncthreads();

  int row = lane >> 1, half = lane & 1;
  float4* zr4 = (float4*)(z + row * DD + half * 128);
  float mean = 0.f, m2 = 0.f;
#pragma unroll 4
  for (int t = 0; t < 32; ++t) {
    float4 v = zr4[t];
    mean += (v.x + v.y) + (v.z + v.w);
  }
  mean += __shfl_xor(mean, 1, 32);
  mean *= (1.f / 256.f);
#pragma unroll 4
  for (int t = 0; t < 32; ++t) {
    float4 v = zr4[t];
    float d0 = v.x - mean, d1 = v.y - mean, d2 = v.z - mean, d3 = v.w - mean;
    m2 += (d0 * d0 + d1 * d1) + (d2 * d2 + d3 * d3);
  }
  m2 += __shfl_xor(m2, 1, 32);
  float inv = rsqrtf(m2 * (1.f / 256.f) + 1e-5f);
  float4* ob4 = (float4*)(out + ((size_t)b * SS + sblk * 16) * DD +
                          row * DD + half * 128);
#pragma unroll 4
  for (int t = 0; t < 32; ++t) {
    float4 v = zr4[t];
    ob4[t] = make_float4((v.x - mean) * inv, (v.y - mean) * inv,
                         (v.z - mean) * inv, (v.w - mean) * inv);
  }
}

// ---------------------------------------------------------------------------
extern "C" void kernel_launch(void* const* d_in, const int* in_sizes, int n_in,
                              void* d_out, int out_size, void* d_ws, size_t ws_size,
                              hipStream_t stream) {
  (void)in_sizes; (void)n_in; (void)out_size; (void)ws_size;
  const float* x    = (const float*)d_in[0];
  const int*   mask = (const int*)d_in[1];
  const float* Wq   = (const float*)d_in[2];
  const float* Wk   = (const float*)d_in[3];
  const float* Wv   = (const float*)d_in[4];
  const float* Wfc  = (const float*)d_in[5];
  const float* W1   = (const float*)d_in[6];
  const float* W2   = (const float*)d_in[7];
  float* out = (float*)d_out;

  char*  wsb = (char*)d_ws;
  size_t off = 0;
  auto take = [&](size_t bytes) -> void* {
    void* p = wsb + off;
    off = (off + bytes + 255) & ~(size_t)255;
    return p;
  };

  const size_t nX   = (size_t)BB * SS * DD;        // 4,194,304
  const size_t nWr  = (size_t)RR * DD * HH;        // 98,304
  const size_t nW12 = (size_t)DD * FF;             // 65,536
  const size_t nQ   = (size_t)BB * RR * SS * HH;   // 6,291,456

  _Float16* xh   = (_Float16*)take(nX   * 2);
  _Float16* wqh  = (_Float16*)take(nWr  * 2);
  _Float16* wkh  = (_Float16*)take(nWr  * 2);
  _Float16* wvh  = (_Float16*)take(nWr  * 2);
  _Float16* wfch = (_Float16*)take(nWr  * 2);
  _Float16* w1h  = (_Float16*)take(nW12 * 2);
  _Float16* w2h  = (_Float16*)take(nW12 * 2);
  _Float16* Qh   = (_Float16*)take(nQ   * 2);
  _Float16* Kth  = (_Float16*)take(nQ   * 2);
  _Float16* Vh   = (_Float16*)take(nQ   * 2);
  _Float16* ctxh = (_Float16*)take(nQ   * 2);
  float*    s32  = (float*)   take(nX   * 4);
  _Float16* shh  = (_Float16*)take(nX   * 2);

  auto cvt = [&](const float* src, _Float16* dst, size_t n) {
    int n4 = (int)(n / 4);
    cvt_f32_f16_v4<<<(unsigned)((n4 + 255) / 256), 256, 0, stream>>>(
        (const float4*)src, (v4h*)dst, n4);
  };
  cvt(x,   xh,   nX);
  cvt(Wq,  wqh,  nWr);
  cvt(Wk,  wkh,  nWr);
  cvt(Wv,  wvh,  nWr);
  cvt(Wfc, wfch, nWr);
  cvt(W1,  w1h,  nW12);
  cvt(W2,  w2h,  nW12);

  const int nBlkBRS = BB * RR * (SS / 16);  // 6144
  const int nBlkBS  = BB * (SS / 16);       // 1024

  qkv_kernel<<<dim3(nBlkBRS, 3), 32, 0, stream>>>(xh, wqh, wkh, wvh, Qh, Kth, Vh);
  attn_kernel<<<nBlkBRS, 32, 0, stream>>>(Qh, Kth, Vh, mask, ctxh);
  proj_kernel<<<nBlkBS, 32, 0, stream>>>(ctxh, wfch, x, s32, shh);
  ffn_kernel<<<nBlkBS, 32, 0, stream>>>(shh, w1h, w2h, s32, out);
}
